// PET2_6210522710124
// MI455X (gfx1250) — matrix operand order repaired and checked by reading, not compiled, not run on hardware
//
#include <hip/hip_runtime.h>
#include <math.h>

// ---------------- model constants ----------------
constexpr int cB  = 16;
constexpr int cN  = 256;
constexpr int cIN = 13;
constexpr int cD  = 128;
constexpr int cH  = 4;
constexpr int cT  = 4;
constexpr int cK  = 15;
constexpr int cDH = 32;
constexpr int cS  = cT + cN;   // 260
constexpr int cHID = 256;      // MLP_R * D
constexpr float cNEG = -1e9f;
constexpr float cEPS = 1e-6f;

// ---------------- WMMA types ----------------
typedef __bf16 bf16_t;
typedef __attribute__((ext_vector_type(16))) __bf16 v16bf;
typedef __attribute__((ext_vector_type(8)))  __bf16 v8bf;
typedef __attribute__((ext_vector_type(8)))  float  v8f;
typedef __attribute__((ext_vector_type(4)))  float  v4f;

__device__ __forceinline__ float gelu_exact(float x) {
  return 0.5f * x * (1.0f + erff(x * 0.7071067811865475f));
}

// Convert 4x contiguous float4 runs into a bf16 fragment (e -> p0..p3 in order).
__device__ __forceinline__ v16bf cvt16(const float* p0, const float* p1,
                                       const float* p2, const float* p3) {
  v4f a0 = *(const v4f*)p0, a1 = *(const v4f*)p1, a2 = *(const v4f*)p2, a3 = *(const v4f*)p3;
  v16bf r;
#pragma unroll
  for (int e = 0; e < 4; ++e) {
    r[e]      = (bf16_t)a0[e];
    r[e + 4]  = (bf16_t)a1[e];
    r[e + 8]  = (bf16_t)a2[e];
    r[e + 12] = (bf16_t)a3[e];
  }
  return r;
}

// A fragment: 16x32 tile of row-major A[M,Kd]. Lane layout (contiguous K order, same as the B loaders):
// lane m%16 -> row, lane/16 selects K halves; element e -> K = k0 + e + 16*half.
// Fast path (tile fully interior, lda%4==0): one contiguous 16-float run -> 4x b128.
__device__ __forceinline__ v16bf load_a(const float* __restrict__ A, int lda,
                                        int row0, int k0, int M, int Kd, int lane) {
  int m = lane & 15, half = lane >> 4;
  int row = row0 + m;
  const float* ap = A + (long)row * lda;
  if (row0 + 16 <= M && k0 + 32 <= Kd && ((lda & 3) == 0)) {
    int base = k0 + 16 * half;
    return cvt16(ap + base, ap + base + 4, ap + base + 8, ap + base + 12);
  }
  bool rok = row < M;
  v16bf r;
#pragma unroll
  for (int e = 0; e < 16; ++e) {
    int k = k0 + e + 16 * half;
    float v = (rok && k < Kd) ? ap[k] : 0.0f;
    r[e] = (bf16_t)v;
  }
  return r;
}

// B fragment from transposed float W[Nout,Kd] (B = W^T): per lane 16 contiguous floats.
__device__ __forceinline__ v16bf load_b_tr(const float* __restrict__ W, int ldw,
                                           int k0, int col0, int Kd, int Nout, int lane) {
  int n = lane & 15, kb = (lane >> 4) * 16;
  int col = col0 + n;
  const float* wp = W + (long)col * ldw + k0 + kb;
  if (col0 + 16 <= Nout && k0 + 32 <= Kd && ((ldw & 3) == 0)) {
    return cvt16(wp, wp + 4, wp + 8, wp + 12);
  }
  v16bf r;
#pragma unroll
  for (int e = 0; e < 16; ++e) {
    int k = k0 + kb + e;
    float v = (k < Kd && col < Nout) ? W[(long)col * ldw + k] : 0.0f;
    r[e] = (bf16_t)v;
  }
  return r;
}

// B fragment from row-major float W[Kd,Nout] (strided; guard hoisted).
__device__ __forceinline__ v16bf load_b_f(const float* __restrict__ W, int ldw,
                                          int k0, int col0, int Kd, int Nout, int lane) {
  int n = lane & 15, kb = (lane >> 4) * 16;
  int col = col0 + n;
  v16bf r;
  if (col0 + 16 <= Nout && k0 + 32 <= Kd) {
#pragma unroll
    for (int e = 0; e < 16; ++e) r[e] = (bf16_t)W[(long)(k0 + kb + e) * ldw + col];
  } else {
#pragma unroll
    for (int e = 0; e < 16; ++e) {
      int k = k0 + kb + e;
      float v = (k < Kd && col < Nout) ? W[(long)k * ldw + col] : 0.0f;
      r[e] = (bf16_t)v;
    }
  }
  return r;
}

// Packed B fragment: one aligned 32-byte vector load.
__device__ __forceinline__ v16bf load_b_p(const bf16_t* __restrict__ Wp, int NtB,
                                          int kt, int nt, int lane) {
  return *(const v16bf*)(Wp + ((((long)kt * NtB + nt) * 32 + lane) << 4));
}

// Combine two v8bf LDS runs into an A fragment.
__device__ __forceinline__ v16bf frag_from_lds(const bf16_t* rowp, int ks, int half) {
  const v8bf* vp = (const v8bf*)rowp;
  v8bf lo = vp[(ks * 32 + 16 * half) >> 3];
  v8bf hi = vp[(ks * 32 + 16 * half + 8) >> 3];
  return __builtin_shufflevector(lo, hi, 0, 1, 2, 3, 4, 5, 6, 7, 8, 9, 10, 11, 12, 13, 14, 15);
}

// ---------------- weight pre-pack: f32 [Kd,Nout] -> bf16 B-fragment tiles ----------------
__global__ void k_pack(const float* __restrict__ W, bf16_t* __restrict__ out,
                       int Kd, int Nout, int NtB) {
  int tile = blockIdx.x;
  int kt = tile / NtB, nt = tile % NtB;
  int lane = threadIdx.x;
  int n = lane & 15, kb = (lane >> 4) * 16;
  int col = nt * 16 + n;
  v16bf r;
#pragma unroll
  for (int e = 0; e < 16; ++e) {
    int k = kt * 32 + kb + e;
    float v = (k < Kd && col < Nout) ? W[(long)k * Nout + col] : 0.0f;
    r[e] = (bf16_t)v;
  }
  *(v16bf*)(out + (((long)tile * 32 + lane) << 4)) = r;
}

// ---------------- generic WMMA GEMM ----------------
// Out = act( alpha * (A @ B) + bias[col] + addm[row,col] )
// BMODE: 0 = float row-major B, 1 = float transposed B, 2 = packed bf16 B.
// block = (32,4): 4 waves, each one 16-col tile. grid.z batching: z -> (b=z/Hdim, h=z%Hdim).
template <int ACT, int BMODE>
__global__ void k_gemm(const float* __restrict__ A, long aSb, long aSh, int lda,
                       const void* __restrict__ Wv, long wSb, long wSh, int ldw, int NtB,
                       const float* __restrict__ bias,
                       const float* __restrict__ addm, long adSb, long adSh, int ldad,
                       float alpha,
                       float* __restrict__ O, long oSb, long oSh, int ldo,
                       int M, int Kd, int Nout, int Hdim) {
  int z = blockIdx.z, bb = z / Hdim, hh = z % Hdim;
  A += (long)bb * aSb + (long)hh * aSh;
  O += (long)bb * oSb + (long)hh * oSh;
  if (addm) addm += (long)bb * adSb + (long)hh * adSh;
  int row0 = blockIdx.x * 16;
  int nt = blockIdx.y * 4 + threadIdx.y;
  int col0 = nt * 16;
  if (col0 >= Nout) return;
  int lane = threadIdx.x;
  v8f acc = {};
  if (BMODE == 2) {
    const bf16_t* Wp = (const bf16_t*)Wv;
    int kt = 0;
    for (int k0 = 0; k0 < Kd; k0 += 32, ++kt) {
      v16bf a = load_a(A, lda, row0, k0, M, Kd, lane);
      v16bf b = load_b_p(Wp, NtB, kt, nt, lane);
      acc = __builtin_amdgcn_wmma_f32_16x16x32_bf16(false, a, false, b, (short)0, acc, false, false);
    }
  } else {
    const float* W = (const float*)Wv + (long)bb * wSb + (long)hh * wSh;
    for (int k0 = 0; k0 < Kd; k0 += 32) {
      v16bf a = load_a(A, lda, row0, k0, M, Kd, lane);
      v16bf b = (BMODE == 1) ? load_b_tr(W, ldw, k0, col0, Kd, Nout, lane)
                             : load_b_f (W, ldw, k0, col0, Kd, Nout, lane);
      acc = __builtin_amdgcn_wmma_f32_16x16x32_bf16(false, a, false, b, (short)0, acc, false, false);
    }
  }
  int n = lane & 15, half = lane >> 4;
  int col = col0 + n;
  if (col < Nout) {
    float bv = bias ? bias[col] : 0.0f;
#pragma unroll
    for (int v = 0; v < 8; ++v) {
      int row = row0 + v + 8 * half;
      if (row < M) {
        float x = alpha * acc[v] + bv;
        if (addm) x += addm[(long)row * ldad + col];
        O[(long)row * ldo + col] = (ACT == 1) ? gelu_exact(x) : x;
      }
    }
  }
}

// ---------------- fused interaction MLP: feats(4) -> 128 (gelu) -> 4 heads ----------------
// block = (32,4): each wave owns a 16-pair row tile; hidden tile in LDS (bf16).
__global__ void k_inter(const float* __restrict__ prep, const float* __restrict__ mf,
                        const float* __restrict__ na, const float* __restrict__ nw,
                        const float* __restrict__ nb,
                        const bf16_t* __restrict__ w1p, const float* __restrict__ b1,
                        const bf16_t* __restrict__ w2p, const float* __restrict__ b2,
                        float* __restrict__ biasB) {
  __shared__ bf16_t lh[4][16][cD];
  int lane = threadIdx.x, ty = threadIdx.y;
  long r0 = ((long)blockIdx.x * 4 + ty) * 16;
  int m = lane & 15, half = lane >> 4;
  long r = r0 + m;
  int b = (int)(r / ((long)cN * cN));
  int rem = (int)(r % ((long)cN * cN));
  int i = rem / cN, j = rem % cN;
  const float* pi = prep + ((long)b * cN + i) * 8;
  const float* pj = prep + ((long)b * cN + j) * 8;
  v4f pia = *(const v4f*)pi, pib = *(const v4f*)(pi + 4);
  v4f pja = *(const v4f*)pj, pjb = *(const v4f*)(pj + 4);
  float deta = pia[0] - pja[0], dphi = pia[1] - pja[1];
  float delta = sqrtf(deta * deta + dphi * dphi + cEPS);
  float pti = pia[2], ptj = pja[2];
  float ptmin = fminf(pti, ptj);
  float kt = ptmin * delta;
  float zf = ptmin / (pti + ptj + cEPS);
  float Es = pia[3] + pja[3], pxs = pib[0] + pjb[0], pys = pib[1] + pjb[1], pzs = pib[2] + pjb[2];
  float m2 = Es * Es - pxs * pxs - pys * pys - pzs * pzs;
  float f[4] = { logf(delta + cEPS), logf(kt + cEPS), logf(zf + cEPS), logf(fmaxf(m2, cEPS)) };
  float av = na[0];
  v16bf afrag;
#pragma unroll
  for (int e = 0; e < 16; ++e) {
    int k = e + 16 * half;
    float v = 0.0f;
    if (k < 4) v = nw[k] * tanhf(av * f[k]) + nb[k];
    afrag[e] = (bf16_t)v;
  }
  // GEMM1: [16x4(pad32)] @ [4x128] -> gelu -> LDS    (w1 packed, NtB=8, kt=0)
  for (int nt = 0; nt < cD / 16; ++nt) {
    v16bf bf = load_b_p(w1p, cD / 16, 0, nt, lane);
    v8f acc = {};
    acc = __builtin_amdgcn_wmma_f32_16x16x32_bf16(false, afrag, false, bf, (short)0, acc, false, false);
    int n = lane & 15, col = nt * 16 + n;
#pragma unroll
    for (int v = 0; v < 8; ++v)
      lh[ty][v + 8 * half][col] = (bf16_t)gelu_exact(acc[v] + b1[col]);
  }
  __syncthreads();
  // GEMM2: [16x128] @ [128x4]   (w2 packed, NtB=1)
  v8f acc2 = {};
  for (int ks = 0; ks < cD / 32; ++ks) {
    v16bf a2 = frag_from_lds(&lh[ty][m][0], ks, half);
    v16bf b2f = load_b_p(w2p, 1, ks, 0, lane);
    acc2 = __builtin_amdgcn_wmma_f32_16x16x32_bf16(false, a2, false, b2f, (short)0, acc2, false, false);
  }
  int n = lane & 15;
  if (n < cH) {
#pragma unroll
    for (int v = 0; v < 8; ++v) {
      long rr = r0 + v + 8 * half;
      int b2i = (int)(rr / ((long)cN * cN));
      int rem2 = (int)(rr % ((long)cN * cN));
      int ii = rem2 / cN, jj = rem2 % cN;
      float pm = mf[b2i * cN + ii] * mf[b2i * cN + jj];
      float val = (acc2[v] + b2[n]) * pm;
      long o = (((long)(b2i * cH + n)) * cS + (cT + ii)) * cS + (cT + jj);
      biasB[o] += val;  // unique writer per element
    }
  }
}

// ---------------- fused local (EdgeConv) MLP: edge(26) -> 256 (gelu) -> 128, mean over K ----------
__global__ void k_local(const float* __restrict__ x, const int* __restrict__ idx,
                        const float* __restrict__ mf,
                        const float* __restrict__ na, const float* __restrict__ nw,
                        const float* __restrict__ nbp,
                        const bf16_t* __restrict__ w1p, const float* __restrict__ b1,
                        const bf16_t* __restrict__ w2p, const float* __restrict__ b2,
                        float* __restrict__ xs) {
  __shared__ bf16_t lh[4][16][cHID];
  int lane = threadIdx.x, ty = threadIdx.y;
  long r0 = ((long)blockIdx.x * 4 + ty) * 16;
  int m = lane & 15, half = lane >> 4;
  long r = r0 + m;
  int b = (int)(r / (cN * cK));
  int rem = (int)(r % (cN * cK));
  int i = rem / cK, kk = rem % cK;
  int j = idx[(b * cN + i) * cK + kk];
  const float* xi = x + ((long)b * cN + i) * cIN;
  const float* xj = x + ((long)b * cN + j) * cIN;
  float av = na[0];
  v16bf afrag;
#pragma unroll
  for (int e = 0; e < 16; ++e) {
    int c = e + 16 * half;
    float v = 0.0f;
    if (c < 2 * cIN) {
      float ev = (c < cIN) ? xi[c] : (xj[c - cIN] - xi[c - cIN]);
      v = nw[c] * tanhf(av * ev) + nbp[c];
    }
    afrag[e] = (bf16_t)v;
  }
  for (int nt = 0; nt < cHID / 16; ++nt) {
    v16bf bf = load_b_p(w1p, cHID / 16, 0, nt, lane);
    v8f acc = {};
    acc = __builtin_amdgcn_wmma_f32_16x16x32_bf16(false, afrag, false, bf, (short)0, acc, false, false);
    int n = lane & 15, col = nt * 16 + n;
#pragma unroll
    for (int v = 0; v < 8; ++v)
      lh[ty][v + 8 * half][col] = (bf16_t)gelu_exact(acc[v] + b1[col]);
  }
  __syncthreads();
  for (int ot = 0; ot < cD / 16; ++ot) {
    v8f acc = {};
    for (int ks = 0; ks < cHID / 32; ++ks) {
      v16bf a2 = frag_from_lds(&lh[ty][m][0], ks, half);
      v16bf b2f = load_b_p(w2p, cD / 16, ks, ot, lane);
      acc = __builtin_amdgcn_wmma_f32_16x16x32_bf16(false, a2, false, b2f, (short)0, acc, false, false);
    }
    int n = lane & 15, col = ot * 16 + n;
#pragma unroll
    for (int v = 0; v < 8; ++v) {
      long rr = r0 + v + 8 * half;
      int bb = (int)(rr / (cN * cK));
      int rem2 = (int)(rr % (cN * cK));
      int ii = rem2 / cK;
      float s = mf[bb * cN + ii] * (1.0f / (float)cK);
      atomicAdd(&xs[(((long)bb * cS) + (cT + ii)) * cD + col], (acc[v] + b2[col]) * s);
    }
  }
}

// ---------------- small elementwise / reduction kernels ----------------
__global__ void k_prep(const float* __restrict__ x, float* __restrict__ mf,
                       float* __restrict__ prep) {
  int t = blockIdx.x * blockDim.x + threadIdx.x;
  if (t >= cB * cN) return;
  const float* xp = x + (long)t * cIN;
  mf[t] = (xp[2] != 0.0f) ? 1.0f : 0.0f;
  float eta = xp[0], phi = xp[1];
  float pt = expf(xp[2]), E = expf(xp[3]);
  float* pr = prep + (long)t * 8;
  pr[0] = eta; pr[1] = phi; pr[2] = pt; pr[3] = E;
  pr[4] = pt * cosf(phi); pr[5] = pt * sinf(phi); pr[6] = pt * sinhf(eta); pr[7] = 0.0f;
}

__global__ void k_knn(const float* __restrict__ x, const float* __restrict__ mf,
                      int* __restrict__ idx) {
  int t = blockIdx.x * blockDim.x + threadIdx.x;
  if (t >= cB * cN) return;
  int b = t / cN, i = t % cN;
  const float* xb = x + (long)b * cN * cIN;
  float si = 999.0f * (1.0f - mf[b * cN + i]);
  float p0 = si + xb[i * cIN + 0], p1 = si + xb[i * cIN + 1], p2 = si + xb[i * cIN + 2];
  float bd[cK]; int bi[cK];
#pragma unroll
  for (int q = 0; q < cK; ++q) { bd[q] = 3.4e38f; bi[q] = 0; }
  for (int j = 0; j < cN; ++j) {
    float sj = 999.0f * (1.0f - mf[b * cN + j]);
    float d0 = p0 - (sj + xb[j * cIN + 0]);
    float d1 = p1 - (sj + xb[j * cIN + 1]);
    float d2 = p2 - (sj + xb[j * cIN + 2]);
    float d = d0 * d0 + d1 * d1 + d2 * d2;
    if (d < bd[cK - 1]) {
      int q = cK - 1;
      while (q > 0 && bd[q - 1] > d) { bd[q] = bd[q - 1]; bi[q] = bi[q - 1]; --q; }
      bd[q] = d; bi[q] = j;
    }
  }
  for (int q = 0; q < cK; ++q) idx[t * cK + q] = bi[q];
}

__global__ void k_dyt(const float* __restrict__ in, float* __restrict__ out,
                      const float* __restrict__ a, const float* __restrict__ w,
                      const float* __restrict__ b, long n, int C) {
  long t = (long)blockIdx.x * blockDim.x + threadIdx.x;
  if (t >= n) return;
  int c = (int)(t % C);
  out[t] = w[c] * tanhf(a[0] * in[t]) + b[c];
}

__global__ void k_xs0(const float* __restrict__ token, const float* __restrict__ xe,
                      const float* __restrict__ mf, float* __restrict__ xs) {
  long t = (long)blockIdx.x * blockDim.x + threadIdx.x;
  if (t >= (long)cB * cS * cD) return;
  int c = (int)(t % cD);
  long rs = t / cD;
  int s = (int)(rs % cS), b = (int)(rs / cS);
  float v;
  if (s < cT) v = token[s * cD + c];
  else        v = xe[((long)b * cN + (s - cT)) * cD + c] * mf[b * cN + s - cT];
  xs[t] = v;
}

__global__ void k_biasinit(const float* __restrict__ mf, float* __restrict__ biasB) {
  long t = (long)blockIdx.x * blockDim.x + threadIdx.x;
  if (t >= (long)cB * cH * cS * cS) return;
  int kc = (int)(t % cS);
  long r = t / cS;
  int q = (int)(r % cS);
  int b = (int)((r / cS) / cH);
  float fq = (q  < cT) ? 1.0f : mf[b * cN + q  - cT];
  float fk = (kc < cT) ? 1.0f : mf[b * cN + kc - cT];
  biasB[t] = (fq * fk > 0.0f) ? 0.0f : cNEG;
}

__global__ void k_softmax(float* __restrict__ P, int L) {
  float* p = P + (long)blockIdx.x * L;
  int lane = threadIdx.x;
  float mx = -3.4e38f;
  for (int c = lane; c < L; c += 32) mx = fmaxf(mx, p[c]);
  for (int o = 16; o > 0; o >>= 1) mx = fmaxf(mx, __shfl_xor(mx, o, 32));
  float sum = 0.0f;
  for (int c = lane; c < L; c += 32) { float e = expf(p[c] - mx); p[c] = e; sum += e; }
  for (int o = 16; o > 0; o >>= 1) sum += __shfl_xor(sum, o, 32);
  float inv = 1.0f / sum;
  for (int c = lane; c < L; c += 32) p[c] *= inv;
}

__global__ void k_dytmask(float* __restrict__ xs, const float* __restrict__ mf,
                          const float* __restrict__ a, const float* __restrict__ w,
                          const float* __restrict__ b) {
  long t = (long)blockIdx.x * blockDim.x + threadIdx.x;
  if (t >= (long)cB * cS * cD) return;
  int c = (int)(t % cD);
  long rs = t / cD;
  int s = (int)(rs % cS), bb = (int)(rs / cS);
  float fmv = (s < cT) ? 1.0f : mf[bb * cN + s - cT];
  xs[t] = (w[c] * tanhf(a[0] * xs[t]) + b[c]) * fmv;
}

__global__ void k_kbias(const float* __restrict__ mf, float* __restrict__ kb) {
  int t = blockIdx.x * blockDim.x + threadIdx.x;
  if (t >= cB * cS) return;
  int s = t % cS, b = t / cS;
  float m = (s < cT) ? 1.0f : mf[b * cN + s - cT];
  kb[t] = (m > 0.0f) ? 0.0f : cNEG;
}

__global__ void k_tokw(const float* __restrict__ tbuf, float* __restrict__ xs) {
  int t = blockIdx.x * blockDim.x + threadIdx.x;
  if (t >= cB * cT * cD) return;
  int f = t % (cT * cD), b = t / (cT * cD);
  xs[(long)b * cS * cD + f] = tbuf[t];
}

__global__ void k_fcn(const float* __restrict__ xs, float* __restrict__ flatn,
                      const float* __restrict__ a, const float* __restrict__ w,
                      const float* __restrict__ b) {
  int t = blockIdx.x * blockDim.x + threadIdx.x;
  if (t >= cB * cT * cD) return;
  int f = t % (cT * cD), bb = t / (cT * cD);
  float v = xs[(long)bb * cS * cD + f];
  flatn[t] = w[f] * tanhf(a[0] * v) + b[f];
}

// ---------------- host side ----------------
static inline int cdiv(long a, int b) { return (int)((a + b - 1) / b); }

template <int ACT, int BMODE>
static inline void gemm(hipStream_t s,
                        const float* A, long aSb, long aSh, int lda,
                        const void* W, long wSb, long wSh, int ldw, int NtB,
                        const float* bias,
                        const float* addm, long adSb, long adSh, int ldad,
                        float alpha,
                        float* O, long oSb, long oSh, int ldo,
                        int M, int Kd, int Nout, int Z, int Hdim) {
  dim3 g(cdiv(M, 16), cdiv(Nout, 64), Z);
  k_gemm<ACT, BMODE><<<g, dim3(32, 4), 0, s>>>(A, aSb, aSh, lda, W, wSb, wSh, ldw, NtB, bias,
                                               addm, adSb, adSh, ldad, alpha,
                                               O, oSb, oSh, ldo, M, Kd, Nout, Hdim);
}

struct Blk { const float *n1a,*n1w,*n1b,*qw,*qb,*pw,*pb,*n2a,*n2w,*n2b,*f1w,*f1b,*f2w,*f2b; };

extern "C" void kernel_launch(void* const* d_in, const int* in_sizes, int n_in,
                              void* d_out, int out_size, void* d_ws, size_t ws_size,
                              hipStream_t stream) {
  (void)in_sizes; (void)n_in; (void)out_size; (void)ws_size;
  int p = 0;
  auto nxt = [&]() { return (const float*)d_in[p++]; };
  const float* X = nxt();
  const float *e_na=nxt(), *e_nw=nxt(), *e_nb=nxt(), *e_f1w=nxt(), *e_f1b=nxt(), *e_f2w=nxt(), *e_f2b=nxt();
  const float *i_na=nxt(), *i_nw=nxt(), *i_nb=nxt(), *i_f1w=nxt(), *i_f1b=nxt(), *i_f2w=nxt(), *i_f2b=nxt();
  const float *l_na=nxt(), *l_nw=nxt(), *l_nb=nxt(), *l_f1w=nxt(), *l_f1b=nxt(), *l_f2w=nxt(), *l_f2b=nxt();
  const float* tok = nxt();
  Blk blk[4];
  for (int z = 0; z < 4; ++z) {
    blk[z].n1a=nxt(); blk[z].n1w=nxt(); blk[z].n1b=nxt();
    blk[z].qw=nxt();  blk[z].qb=nxt();
    blk[z].pw=nxt();  blk[z].pb=nxt();
    blk[z].n2a=nxt(); blk[z].n2w=nxt(); blk[z].n2b=nxt();
    blk[z].f1w=nxt(); blk[z].f1b=nxt(); blk[z].f2w=nxt(); blk[z].f2b=nxt();
  }
  const float *nrm_a=nxt(), *nrm_w=nxt(), *nrm_b=nxt();
  const float *fc_na=nxt(), *fc_nw=nxt(), *fc_nb=nxt(), *fc_f1w=nxt(), *fc_f1b=nxt(), *fc_f2w=nxt(), *fc_f2b=nxt();
  const float *o_w=nxt(), *o_b=nxt();

  // workspace layout (floats, 64B aligned blocks)
  float* W0 = (float*)d_ws;
  size_t off = 0;
  auto alloc = [&](size_t n) { off = (off + 15) & ~(size_t)15; float* r = W0 + off; off += n; return r; };
  float* mf    = alloc((size_t)cB * cN);
  int*   idx   = (int*)alloc((size_t)cB * cN * cK);
  float* prep  = alloc((size_t)cB * cN * 8);
  float* xe    = alloc((size_t)cB * cN * cD);
  float* xs    = alloc((size_t)cB * cS * cD);
  float* xsn   = alloc((size_t)cB * cS * cD);
  float* qkv   = alloc((size_t)cB * cS * 3 * cD);
  float* obuf  = alloc((size_t)cB * cS * cD);
  float* h1    = alloc((size_t)cB * cS * cHID);
  float* biasB = alloc((size_t)cB * cH * cS * cS);
  float* probs = alloc((size_t)cB * cH * cS * cS);
  float* kb    = alloc((size_t)cB * cS);
  float* tbuf  = alloc((size_t)cB * cT * cD);
  float* tn    = alloc((size_t)cB * cT * cD);
  float* th    = alloc((size_t)cB * cT * cHID);
  float* e13   = alloc((size_t)cB * cN * cIN);
  float* flatn = alloc((size_t)cB * cT * cD);
  float* fch   = alloc((size_t)cB * 2 * cT * cD);
  float* fco   = alloc((size_t)cB * cT * cD);

  // ---- pre-pack weights to bf16 fragment layout ----
  auto pack = [&](const float* Wm, int Kd, int Nout) -> bf16_t* {
    int kt = (Kd + 31) / 32, nt = (Nout + 15) / 16;
    bf16_t* dst = (bf16_t*)alloc((size_t)kt * nt * 256);  // 512 bf16 per tile
    k_pack<<<dim3(kt * nt), dim3(32), 0, stream>>>(Wm, dst, Kd, Nout, nt);
    return dst;
  };
  bf16_t* e_f1p = pack(e_f1w, cIN, cHID);
  bf16_t* e_f2p = pack(e_f2w, cHID, cD);
  bf16_t* i_f1p = pack(i_f1w, 4, cD);
  bf16_t* i_f2p = pack(i_f2w, cD, cH);
  bf16_t* l_f1p = pack(l_f1w, 2 * cIN, cHID);
  bf16_t* l_f2p = pack(l_f2w, cHID, cD);
  bf16_t *qwp[4], *pwp[4], *f1p[4], *f2p[4];
  for (int z = 0; z < 4; ++z) {
    qwp[z] = pack(blk[z].qw, cD, 3 * cD);
    pwp[z] = pack(blk[z].pw, cD, cD);
    f1p[z] = pack(blk[z].f1w, cD, cHID);
    f2p[z] = pack(blk[z].f2w, cHID, cD);
  }
  bf16_t* fc1p = pack(fc_f1w, cT * cD, 2 * cT * cD);
  bf16_t* fc2p = pack(fc_f2w, 2 * cT * cD, cT * cD);
  bf16_t* owp  = pack(o_w, cT * cD, 2);

  const float ascale = 1.0f / sqrtf((float)cDH);
  const int EB = 256;

  // particle features + mask, KNN
  k_prep<<<cdiv(cB * cN, EB), EB, 0, stream>>>(X, mf, prep);
  k_knn<<<cdiv(cB * cN, 128), 128, 0, stream>>>(X, mf, idx);

  // InputBlock embed: dyt -> 13->256 gelu -> 256->128
  k_dyt<<<cdiv((long)cB * cN * cIN, EB), EB, 0, stream>>>(X, e13, e_na, e_nw, e_nb, (long)cB * cN * cIN, cIN);
  gemm<1,2>(stream, e13,0,0,cIN, e_f1p,0,0,0,cHID/16, e_f1b, nullptr,0,0,0, 1.0f, h1,0,0,cHID, cB*cN, cIN, cHID, 1,1);
  gemm<0,2>(stream, h1,0,0,cHID, e_f2p,0,0,0,cD/16, e_f2b, nullptr,0,0,0, 1.0f, xe,0,0,cD, cB*cN, cHID, cD, 1,1);

  // assemble xs = [tokens | xe*mf]; bias init; then fused local + interaction add in
  k_xs0<<<cdiv((long)cB * cS * cD, EB), EB, 0, stream>>>(tok, xe, mf, xs);
  k_biasinit<<<cdiv((long)cB * cH * cS * cS, EB), EB, 0, stream>>>(mf, biasB);
  k_local<<<dim3(cB * cN * cK / 64), dim3(32, 4), 0, stream>>>(X, idx, mf, l_na, l_nw, l_nb,
                                                               l_f1p, l_f1b, l_f2p, l_f2b, xs);
  k_inter<<<dim3(cB * cN * cN / 64), dim3(32, 4), 0, stream>>>(prep, mf, i_na, i_nw, i_nb,
                                                               i_f1p, i_f1b, i_f2p, i_f2b, biasB);

  // two full attention blocks
  for (int z = 0; z < 2; ++z) {
    const Blk& bk = blk[z];
    k_dyt<<<cdiv((long)cB * cS * cD, EB), EB, 0, stream>>>(xs, xsn, bk.n1a, bk.n1w, bk.n1b, (long)cB * cS * cD, cD);
    gemm<0,2>(stream, xsn,0,0,cD, qwp[z],0,0,0,3*cD/16, bk.qb, nullptr,0,0,0, 1.0f, qkv,0,0,3*cD, cB*cS, cD, 3*cD, 1,1);
    // scores = Q K^T / sqrt(DH) + bias  -> probs
    gemm<0,1>(stream, qkv,(long)cS*3*cD,(long)cDH,3*cD,
              qkv + cD,(long)cS*3*cD,(long)cDH,3*cD,0,
              nullptr,
              biasB,(long)cH*cS*cS,(long)cS*cS,cS,
              ascale,
              probs,(long)cH*cS*cS,(long)cS*cS,cS,
              cS, cDH, cS, cB*cH, cH);
    k_softmax<<<dim3(cB * cH * cS), dim3(32), 0, stream>>>(probs, cS);
    // o = probs @ V
    gemm<0,0>(stream, probs,(long)cH*cS*cS,(long)cS*cS,cS,
              qkv + 2*cD,(long)cS*3*cD,(long)cDH,3*cD,0,
              nullptr, nullptr,0,0,0, 1.0f,
              obuf,(long)cS*cD,(long)cDH,cD,
              cS, cS, cDH, cB*cH, cH);
    // xs += o @ proj + b
    gemm<0,2>(stream, obuf,0,0,cD, pwp[z],0,0,0,cD/16, bk.pb, xs,0,0,cD, 1.0f, xs,0,0,cD, cB*cS, cD, cD, 1,1);
    // xs += mlp(dyt(xs))
    k_dyt<<<cdiv((long)cB * cS * cD, EB), EB, 0, stream>>>(xs, xsn, bk.n2a, bk.n2w, bk.n2b, (long)cB * cS * cD, cD);
    gemm<1,2>(stream, xsn,0,0,cD, f1p[z],0,0,0,cHID/16, bk.f1b, nullptr,0,0,0, 1.0f, h1,0,0,cHID, cB*cS, cD, cHID, 1,1);
    gemm<0,2>(stream, h1,0,0,cHID, f2p[z],0,0,0,cD/16, bk.f2b, xs,0,0,cD, 1.0f, xs,0,0,cD, cB*cS, cHID, cD, 1,1);
  }

  // final norm * mask, classifier key bias
  k_dytmask<<<cdiv((long)cB * cS * cD, EB), EB, 0, stream>>>(xs, mf, nrm_a, nrm_w, nrm_b);
  k_kbias<<<cdiv(cB * cS, EB), EB, 0, stream>>>(mf, kb);

  // two class-attention blocks (queries = T tokens)
  for (int z = 2; z < 4; ++z) {
    const Blk& bk = blk[z];
    k_dyt<<<cdiv((long)cB * cS * cD, EB), EB, 0, stream>>>(xs, xsn, bk.n1a, bk.n1w, bk.n1b, (long)cB * cS * cD, cD);
    gemm<0,2>(stream, xsn,0,0,cD, qwp[z],0,0,0,3*cD/16, bk.qb, nullptr,0,0,0, 1.0f, qkv,0,0,3*cD, cB*cS, cD, 3*cD, 1,1);
    // scores [T,S]
    gemm<0,1>(stream, qkv,(long)cS*3*cD,(long)cDH,3*cD,
              qkv + cD,(long)cS*3*cD,(long)cDH,3*cD,0,
              nullptr,
              kb,(long)cS,0,0,
              ascale,
              probs,(long)cH*cT*cS,(long)cT*cS,cS,
              cT, cDH, cS, cB*cH, cH);
    k_softmax<<<dim3(cB * cH * cT), dim3(32), 0, stream>>>(probs, cS);
    gemm<0,0>(stream, probs,(long)cH*cT*cS,(long)cT*cS,cS,
              qkv + 2*cD,(long)cS*3*cD,(long)cDH,3*cD,0,
              nullptr, nullptr,0,0,0, 1.0f,
              obuf,(long)cT*cD,(long)cDH,cD,
              cT, cS, cDH, cB*cH, cH);
    // t = xs[:, :T] + o @ proj + b   (batched over b to index strided xs rows)
    gemm<0,2>(stream, obuf,(long)cT*cD,0,cD, pwp[z],0,0,0,cD/16, bk.pb,
              xs,(long)cS*cD,0,cD, 1.0f,
              tbuf,(long)cT*cD,0,cD, cT, cD, cD, cB, 1);
    // t += mlp(dyt(t))
    k_dyt<<<cdiv((long)cB * cT * cD, EB), EB, 0, stream>>>(tbuf, tn, bk.n2a, bk.n2w, bk.n2b, (long)cB * cT * cD, cD);
    gemm<1,2>(stream, tn,0,0,cD, f1p[z],0,0,0,cHID/16, bk.f1b, nullptr,0,0,0, 1.0f, th,0,0,cHID, cB*cT, cD, cHID, 1,1);
    gemm<0,2>(stream, th,0,0,cHID, f2p[z],0,0,0,cD/16, bk.f2b, tbuf,0,0,cD, 1.0f, tbuf,0,0,cD, cB*cT, cHID, cD, 1,1);
    k_tokw<<<cdiv(cB * cT * cD, EB), EB, 0, stream>>>(tbuf, xs);
  }

  // head: flatten tokens -> dyt -> 512->1024 gelu -> 1024->512 -> 512->2
  k_fcn<<<cdiv(cB * cT * cD, EB), EB, 0, stream>>>(xs, flatn, fc_na, fc_nw, fc_nb);
  gemm<1,2>(stream, flatn,0,0,cT*cD, fc1p,0,0,0,2*cT*cD/16, fc_f1b, nullptr,0,0,0, 1.0f, fch,0,0,2*cT*cD, cB, cT*cD, 2*cT*cD, 1,1);
  gemm<0,2>(stream, fch,0,0,2*cT*cD, fc2p,0,0,0,cT*cD/16, fc_f2b, nullptr,0,0,0, 1.0f, fco,0,0,cT*cD, cB, 2*cT*cD, cT*cD, 1,1);
  gemm<0,2>(stream, fco,0,0,cT*cD, owp,0,0,0,1, o_b, nullptr,0,0,0, 1.0f, (float*)d_out,0,0,2, cB, cT*cD, 2, 1,1);
}
